// IntraViewRegularization_34196529611077
// MI455X (gfx1250) — matrix operand order repaired and checked by reading, not compiled, not run on hardware
//
#include <hip/hip_runtime.h>
#include <math.h>

typedef __attribute__((ext_vector_type(16))) _Float16 v16h;
typedef __attribute__((ext_vector_type(8)))  float    v8f;

#define NCLS 48

// ---------------------------------------------------------------------------
// WMMA helper: D = A(16x32 f16) * B(32x16 f16) + C(16x16 f32)
// ---------------------------------------------------------------------------
__device__ __forceinline__ v8f wmma_acc(v16h a, v16h b, v8f c) {
  return __builtin_amdgcn_wmma_f32_16x16x32_f16(
      /*neg_a=*/false, a, /*neg_b=*/false, b,
      /*c_mod=*/(short)0, c, /*reuse_a=*/false, /*reuse_b=*/false);
}

// Load 8 consecutive classes as p = s * exp(x - lse) into elements O..O+7.
template <int O>
__device__ __forceinline__ void loadA8(v16h& a, const float* p, float lse, float s) {
  const float4* q = reinterpret_cast<const float4*>(p);
  float4 q0 = q[0], q1 = q[1];
  a[O + 0] = (_Float16)(s * __expf(q0.x - lse));
  a[O + 1] = (_Float16)(s * __expf(q0.y - lse));
  a[O + 2] = (_Float16)(s * __expf(q0.z - lse));
  a[O + 3] = (_Float16)(s * __expf(q0.w - lse));
  a[O + 4] = (_Float16)(s * __expf(q1.x - lse));
  a[O + 5] = (_Float16)(s * __expf(q1.y - lse));
  a[O + 6] = (_Float16)(s * __expf(q1.z - lse));
  a[O + 7] = (_Float16)(s * __expf(q1.w - lse));
}

template <int O>
__device__ __forceinline__ void setZero8(v16h& a) {
#pragma unroll
  for (int i = 0; i < 8; ++i) a[O + i] = (_Float16)0.0f;
}

// Load 16 consecutive classes as logp = x - lse into elements 0..15.
__device__ __forceinline__ void loadB16(v16h& b, const float* p, float lse, bool pred) {
  if (pred) {
    const float4* q = reinterpret_cast<const float4*>(p);
    float4 q0 = q[0], q1 = q[1], q2 = q[2], q3 = q[3];
    b[0]  = (_Float16)(q0.x - lse); b[1]  = (_Float16)(q0.y - lse);
    b[2]  = (_Float16)(q0.z - lse); b[3]  = (_Float16)(q0.w - lse);
    b[4]  = (_Float16)(q1.x - lse); b[5]  = (_Float16)(q1.y - lse);
    b[6]  = (_Float16)(q1.z - lse); b[7]  = (_Float16)(q1.w - lse);
    b[8]  = (_Float16)(q2.x - lse); b[9]  = (_Float16)(q2.y - lse);
    b[10] = (_Float16)(q2.z - lse); b[11] = (_Float16)(q2.w - lse);
    b[12] = (_Float16)(q3.x - lse); b[13] = (_Float16)(q3.y - lse);
    b[14] = (_Float16)(q3.z - lse); b[15] = (_Float16)(q3.w - lse);
  } else {
    v16h z = {};
    b = z;
  }
}

// ---------------------------------------------------------------------------
// K0: zero scratch
// ---------------------------------------------------------------------------
__global__ void k_init(float* colsum, int* mask, float* acc, int* count, int N) {
  int i = blockIdx.x * blockDim.x + threadIdx.x;
  if (i < N) { colsum[i] = 0.0f; mask[i] = 0; }
  if (i < 8) acc[i] = 0.0f;
  if (i == 0) *count = 0;
}

// K1: labeled-node mask
__global__ void k_mask(const int* __restrict__ lbl, int* __restrict__ mask, int L) {
  int i = blockIdx.x * blockDim.x + threadIdx.x;
  if (i < L) mask[lbl[i]] = 1;
}

// ---------------------------------------------------------------------------
// K2: per-node logsumexp + negentropy (Σ p log p) for both branches
// ---------------------------------------------------------------------------
__device__ __forceinline__ void node_stats(const float* __restrict__ p,
                                           float& lse_out, float& neg_out) {
  const float4* q = reinterpret_cast<const float4*>(p);
  float v[NCLS];
#pragma unroll
  for (int i = 0; i < NCLS / 4; ++i) {
    float4 t = q[i];
    v[4 * i + 0] = t.x; v[4 * i + 1] = t.y;
    v[4 * i + 2] = t.z; v[4 * i + 3] = t.w;
  }
  float mx = v[0];
#pragma unroll
  for (int i = 1; i < NCLS; ++i) mx = fmaxf(mx, v[i]);
  float S = 0.0f, T = 0.0f;
#pragma unroll
  for (int i = 0; i < NCLS; ++i) {
    float t = v[i] - mx;
    float e = __expf(t);
    S += e;
    T += t * e;
  }
  float ls = __logf(S);
  lse_out = ls + mx;        // log_softmax = x - lse
  neg_out = T / S - ls;     // Σ p * log p
}

__global__ void k_stats(const float* __restrict__ x1, const float* __restrict__ x2,
                        float* __restrict__ lse1, float* __restrict__ neg1,
                        float* __restrict__ lse2, float* __restrict__ neg2, int N) {
  int n = blockIdx.x * blockDim.x + threadIdx.x;
  if (n >= N) return;
  float l, g;
  node_stats(x1 + (size_t)n * NCLS, l, g); lse1[n] = l; neg1[n] = g;
  node_stats(x2 + (size_t)n * NCLS, l, g); lse2[n] = l; neg2[n] = g;
}

// ---------------------------------------------------------------------------
// K3: colsum scatter-add + wave-aggregated compaction of masked edges
// ---------------------------------------------------------------------------
__global__ void k_edges(const int* __restrict__ cols, const float* __restrict__ ew,
                        const int* __restrict__ mask, float* __restrict__ colsum,
                        int* __restrict__ list, int* __restrict__ count, int E) {
  int tid = blockIdx.x * blockDim.x + threadIdx.x;
  int stride = gridDim.x * blockDim.x;
  int lane = threadIdx.x & 31;
  for (int e = tid; e < E; e += stride) {
    int c = cols[e];
    atomicAdd(&colsum[c], ew[e]);
    bool m = mask[c] != 0;
    unsigned mb = (unsigned)__ballot(m);   // wave32: low 32 bits
    if (mb) {
      int leader = __ffs(mb) - 1;
      int tot = __popc(mb);
      int pre = __popc(mb & ((1u << lane) - 1u));
      int base = 0;
      if (lane == leader) base = atomicAdd(count, tot);
      base = __shfl(base, leader, 32);
      if (m) list[base + pre] = e;
    }
  }
}

// ---------------------------------------------------------------------------
// K4: masked-edge pass. 16 edges per wave; per-edge 48-class dot products
//     done as two V_WMMA_F32_16X16X32_F16 (K=32 + K=16 zero-padded); the
//     diagonal of the persistent f32 accumulator holds the dots.
// A layout (16-bit 16x32): lanes 0-15 hold K=0-15; lanes 16-31 K=16-31, as loaded here.
// B layout (16-bit 32x16): lanes 0-15 hold K=0-15; lanes 16-31 K=16-31.
// ---------------------------------------------------------------------------
__global__ void k_kl(const float* __restrict__ x1, const float* __restrict__ x2,
                     const int* __restrict__ rows, const int* __restrict__ cols,
                     const float* __restrict__ ew, const float* __restrict__ colsum,
                     const float* __restrict__ lse1, const float* __restrict__ neg1,
                     const float* __restrict__ lse2, const float* __restrict__ neg2,
                     const int* __restrict__ list, const int* __restrict__ countp,
                     float* __restrict__ acc) {
  const int lane = threadIdx.x & 31;
  const int wid = blockIdx.x * (blockDim.x >> 5) + (threadIdx.x >> 5);
  const int nw = gridDim.x * (blockDim.x >> 5);
  const int count = *countp;
  const bool hi = lane >= 16;
  const int sub = lane & 15;

  v8f acc1 = {};   // branch-1 dot accumulator (diagonal = Σ dots)
  v8f acc2 = {};   // branch-2
  float sw = 0.0f, ns1 = 0.0f, ns2 = 0.0f;

  const int a1c0 = hi ? 16 : 0;   // A wmma#1: classes a1c0..a1c0+15
  const int a2c0 = 32;            // A wmma#2: lo lanes classes 32..47, hi lanes zero-pad
  const int bc0  = hi ? 16 : 0;   // B wmma#1: classes bc0..bc0+15

  for (int base = wid * 16; base < count; base += nw * 16) {  // wave-uniform
    int ei = base + sub;
    bool valid = ei < count;
    int e = valid ? list[ei] : 0;
    int r = rows[e];
    int c = cols[e];

    if (valid && !hi) {  // scalar side terms, one lane per edge
      float we = ew[e];
      float cs = colsum[c];
      sw  += (cs > 0.0f) ? (we / cs) : 0.0f;
      ns1 += neg1[r];
      ns2 += neg2[r];
    }
    float s = valid ? 1.0f : 0.0f;  // zero A rows of padding edges

    {  // branch 1
      float lr = lse1[r], lc = lse1[c];
      const float* xr = x1 + (size_t)r * NCLS;
      const float* xc = x1 + (size_t)c * NCLS;
      v16h a1, a2, b1, b2;
      loadA8<0>(a1, xr + a1c0, lr, s);
      loadA8<8>(a1, xr + a1c0 + 8, lr, s);
      loadA8<0>(a2, xr + a2c0, lr, hi ? 0.0f : s);
      loadA8<8>(a2, xr + a2c0 + 8, lr, hi ? 0.0f : s);
      loadB16(b1, xc + bc0, lc, true);
      loadB16(b2, xc + 32, lc, !hi);         // hi lanes = K 48..63 pad
      acc1 = wmma_acc(a1, b1, acc1);
      acc1 = wmma_acc(a2, b2, acc1);
    }
    {  // branch 2
      float lr = lse2[r], lc = lse2[c];
      const float* xr = x2 + (size_t)r * NCLS;
      const float* xc = x2 + (size_t)c * NCLS;
      v16h a1, a2, b1, b2;
      loadA8<0>(a1, xr + a1c0, lr, s);
      loadA8<8>(a1, xr + a1c0 + 8, lr, s);
      loadA8<0>(a2, xr + a2c0, lr, hi ? 0.0f : s);
      loadA8<8>(a2, xr + a2c0 + 8, lr, hi ? 0.0f : s);
      loadB16(b1, xc + bc0, lc, true);
      loadB16(b2, xc + 32, lc, !hi);
      acc2 = wmma_acc(a1, b1, acc2);
      acc2 = wmma_acc(a2, b2, acc2);
    }
  }

  // Extract diagonal of C: D[m,m] -> (VGPR m, lane m) for m<8,
  //                                  (VGPR m-8, lane m+16) for m>=8.
  float d1 = 0.0f, d2 = 0.0f;
#pragma unroll
  for (int v = 0; v < 8; ++v) {
    bool sel = (lane == v) || (lane == v + 24);
    if (sel) { d1 = acc1[v]; d2 = acc2[v]; }
  }

  float r0 = sw, r1 = ns1 - d1, r2 = ns2 - d2;  // KL = negent[row] - dot
#pragma unroll
  for (int off = 16; off; off >>= 1) {
    r0 += __shfl_xor(r0, off, 32);
    r1 += __shfl_xor(r1, off, 32);
    r2 += __shfl_xor(r2, off, 32);
  }
  if (lane == 0) {
    atomicAdd(&acc[0], r0);
    atomicAdd(&acc[1], r1);
    atomicAdd(&acc[2], r2);
  }
}

// K5: finalize scalar
__global__ void k_fin(const float* __restrict__ acc, const int* __restrict__ countp,
                      float* __restrict__ out, int L) {
  if (threadIdx.x == 0 && blockIdx.x == 0) {
    float nf = (float)(*countp);
    out[0] = acc[0] * (acc[1] + acc[2]) / nf / (float)L;
  }
}

// ---------------------------------------------------------------------------
extern "C" void kernel_launch(void* const* d_in, const int* in_sizes, int n_in,
                              void* d_out, int out_size, void* d_ws, size_t ws_size,
                              hipStream_t stream) {
  (void)n_in; (void)out_size; (void)ws_size;
  const int N = in_sizes[0] / NCLS;   // 50000
  const int E = in_sizes[3];          // 1,600,000
  const int L = in_sizes[4];          // 5000

  const float* x1   = (const float*)d_in[0];
  const float* x2   = (const float*)d_in[1];
  const int*   eidx = (const int*)d_in[2];
  const int*   rows = eidx;           // edge_indices[0]
  const int*   colv = eidx + E;       // edge_indices[1]
  const float* ew   = (const float*)d_in[3];
  const int*   lbl  = (const int*)d_in[4];
  float*       out  = (float*)d_out;

  char* p = (char*)d_ws;
  auto take = [&](size_t bytes) -> char* {
    char* r = p;
    p += (bytes + 255) & ~(size_t)255;
    return r;
  };
  float* colsum = (float*)take((size_t)N * 4);
  float* lse1   = (float*)take((size_t)N * 4);
  float* neg1   = (float*)take((size_t)N * 4);
  float* lse2   = (float*)take((size_t)N * 4);
  float* neg2   = (float*)take((size_t)N * 4);
  int*   mask   = (int*)take((size_t)N * 4);
  float* acc    = (float*)take(256);
  int*   count  = (int*)take(256);
  int*   list   = (int*)take((size_t)E * 4);

  k_init<<<(N + 255) / 256, 256, 0, stream>>>(colsum, mask, acc, count, N);
  k_mask<<<(L + 255) / 256, 256, 0, stream>>>(lbl, mask, L);
  k_stats<<<(N + 255) / 256, 256, 0, stream>>>(x1, x2, lse1, neg1, lse2, neg2, N);

  int eg = (E + 255) / 256;
  if (eg > 4096) eg = 4096;
  k_edges<<<eg, 256, 0, stream>>>(colv, ew, mask, colsum, list, count, E);

  k_kl<<<1024, 256, 0, stream>>>(x1, x2, rows, colv, ew, colsum,
                                 lse1, neg1, lse2, neg2, list, count, acc);
  k_fin<<<1, 32, 0, stream>>>(acc, count, out, L);
}
